// Phi3Attention_19018115186726
// MI455X (gfx1250) — matrix operand order repaired and checked by reading, not compiled, not run on hardware
//
#include <hip/hip_runtime.h>

// ---------------- problem constants ----------------
#define BB   2
#define SS   2048
#define HID  3072
#define NH   32
#define HD   96
#define NHHD (NH * HD)        // 3072
#define OP   (NHHD * 3)       // 9216

typedef __attribute__((ext_vector_type(16))) __bf16 v16bf;
typedef __attribute__((ext_vector_type(8)))  __bf16 v8bf;
typedef __attribute__((ext_vector_type(8)))  float  v8f;

__device__ __forceinline__ v16bf cat16(v8bf lo, v8bf hi) {
  v16bf r;
#pragma unroll
  for (int i = 0; i < 8; ++i) { r[i] = lo[i]; r[i + 8] = hi[i]; }
  return r;
}

// A-fragment (16x32 bf16), row-major source with leading dim ld.
// lanes 0-15: row m0+l, K = {k0..k0+7, k0+16..k0+23}
// lanes 16-31: row m0+l, K = {k0+8..k0+15, k0+24..k0+31}
__device__ __forceinline__ v16bf load_frag_a(const __bf16* base, int ld,
                                             int m0, int k0, int lane) {
  int half = lane >> 4, l = lane & 15;
  const __bf16* p = base + (long)(m0 + l) * ld + k0 + half * 8;
  v8bf lo = *(const v8bf*)(p);
  v8bf hi = *(const v8bf*)(p + 16);
  return cat16(lo, hi);
}

// B-fragment (32x16 bf16) from B^T stored row-major [N, K] with leading dim ld.
// lanes 0-15: col n0+l, K = {k0..k0+7, k0+16..k0+23} ; lanes 16-31: col n0+l, K = {k0+8..k0+15, k0+24..k0+31}
__device__ __forceinline__ v16bf load_frag_b(const __bf16* base, int ld,
                                             int n0, int k0, int lane) {
  int half = lane >> 4, l = lane & 15;
  const __bf16* p = base + (long)(n0 + l) * ld + k0 + half * 8;
  v8bf lo = *(const v8bf*)(p);
  v8bf hi = *(const v8bf*)(p + 16);
  return cat16(lo, hi);
}

__device__ __forceinline__ v8f wmma_bf16(v16bf a, v16bf b, v8f c) {
  return __builtin_amdgcn_wmma_f32_16x16x32_bf16(false, a, false, b,
                                                 (short)0, c, false, false);
}

// Low 32 bits of a generic pointer to LDS == addrspace(3) byte offset.
__device__ __forceinline__ unsigned int lds_off(const void* p) {
  return (unsigned int)(unsigned long long)(uintptr_t)p;
}

__device__ __forceinline__ void async_ld16(unsigned int lds_dst,
                                           const void* gsrc) {
  asm volatile("global_load_async_to_lds_b128 %0, %1, off"
               :: "v"(lds_dst), "v"((unsigned long long)(uintptr_t)gsrc)
               : "memory");
}

// ---------------- conversion kernels ----------------
__global__ void cvt_f32_bf16_kernel(const float* __restrict__ in,
                                    __bf16* __restrict__ out, long n) {
  long i = (long)blockIdx.x * blockDim.x + threadIdx.x;
  if (i < n) out[i] = (__bf16)in[i];
}

// in [R][C] f32  ->  out [C][R] bf16
__global__ void transpose_f32_bf16_kernel(const float* __restrict__ in,
                                          __bf16* __restrict__ out,
                                          int R, int C) {
  long i = (long)blockIdx.x * blockDim.x + threadIdx.x;
  if (i < (long)R * C) {
    int c = (int)(i % C);
    int r = (int)(i / C);
    out[(long)c * R + r] = (__bf16)in[i];
  }
}

// ---------------- generic bf16 GEMM: C[M,N] = A[M,K] * B[K,N] ----------------
// A row-major bf16, Bt = B^T row-major [N,K] bf16.  OUT_F32 selects C dtype.
// Block = 4 waves covering 64 M-rows x 64 N-cols; the 64x32 B k-slice is
// shared by all 4 waves and is double-buffer staged into LDS with the CDNA5
// async global->LDS engine (ASYNCcnt), overlapped with WMMA on the live buffer.
template <int OUT_F32>
__global__ __launch_bounds__(128) void gemm_bf16_kernel(
    const __bf16* __restrict__ A, const __bf16* __restrict__ Bt,
    void* __restrict__ Cout, int M, int N, int K) {
  __shared__ __bf16 btile[2][64][32];  // 2 x 4KB, [n][k] within tile

  int lane = threadIdx.x & 31;
  int wave = threadIdx.x >> 5;
  int tid = threadIdx.x;
  int m0 = blockIdx.y * 64 + wave * 16;
  int n0 = blockIdx.x * 64;

  // Stage the 64(n) x 32(k) bf16 tile at column k0 into btile[buf].
  // 256 x 16B segments; each of 128 threads issues 2 async 16B transfers.
  auto stage = [&](int buf, int k0) {
#pragma unroll
    for (int it = 0; it < 2; ++it) {
      int s = tid + it * 128;          // 0..255
      int n = s >> 2, kseg = s & 3;
      async_ld16(lds_off(&btile[buf][n][kseg * 8]),
                 Bt + (long)(n0 + n) * K + k0 + kseg * 8);
    }
  };

  v8f acc[4];
#pragma unroll
  for (int t = 0; t < 4; ++t)
#pragma unroll
    for (int r = 0; r < 8; ++r) acc[t][r] = 0.0f;

  stage(0, 0);
  asm volatile("s_wait_asynccnt 0x0" ::: "memory");
  __syncthreads();

  int nk = K >> 5;
  for (int j = 0; j < nk; ++j) {
    int k0 = j << 5;
    if (j + 1 < nk) stage((j + 1) & 1, k0 + 32);  // prefetch next k-slice

    v16bf a = load_frag_a(A, K, m0, k0, lane);
    const __bf16* bb = &btile[j & 1][0][0];
#pragma unroll
    for (int t = 0; t < 4; ++t) {
      v16bf b = load_frag_b(bb, 32, t * 16, 0, lane);
      acc[t] = wmma_bf16(a, b, acc[t]);
    }

    asm volatile("s_wait_asynccnt 0x0" ::: "memory");
    __syncthreads();
  }

  int half = lane >> 4, l = lane & 15;
#pragma unroll
  for (int t = 0; t < 4; ++t)
#pragma unroll
    for (int r = 0; r < 8; ++r) {
      long idx = (long)(m0 + r + 8 * half) * N + (n0 + t * 16 + l);
      if (OUT_F32) ((float*)Cout)[idx] = acc[t][r];
      else         ((__bf16*)Cout)[idx] = (__bf16)acc[t][r];
    }
}

// ---------------- RoPE + layout kernel ----------------
// qkv bf16 [B*S, OP] -> Q,K roped bf16 [B,H,S,HD]; V transposed bf16 [B,H,HD,S]
__global__ void rope_kernel(const __bf16* __restrict__ qkv,
                            const float* __restrict__ cosT,
                            const float* __restrict__ sinT,
                            __bf16* __restrict__ Qo, __bf16* __restrict__ Ko,
                            __bf16* __restrict__ Vt) {
  long i = (long)blockIdx.x * blockDim.x + threadIdx.x;
  if (i >= (long)BB * SS * NH * HD) return;
  int d = (int)(i % HD);
  int h = (int)((i / HD) % NH);
  int s = (int)((i / (HD * NH)) % SS);
  int b = (int)(i / ((long)HD * NH * SS));

  long base = ((long)b * SS + s) * OP;
  int qi = h * HD + d;
  float qv = (float)qkv[base + qi];
  float kv = (float)qkv[base + NHHD + qi];
  float vv = (float)qkv[base + 2 * NHHD + qi];

  float c  = cosT[s * HD + d];
  float sn = sinT[s * HD + d];
  int   dro = (d < HD / 2) ? d + HD / 2 : d - HD / 2;
  float sgn = (d < HD / 2) ? -1.0f : 1.0f;
  float qr = sgn * (float)qkv[base + h * HD + dro];
  float kr = sgn * (float)qkv[base + NHHD + h * HD + dro];

  long bh = (long)b * NH + h;
  long qko = (bh * SS + s) * HD + d;
  Qo[qko] = (__bf16)(qv * c + qr * sn);
  Ko[qko] = (__bf16)(kv * c + kr * sn);
  Vt[(bh * HD + d) * SS + s] = (__bf16)vv;
}

// ---------------- flash attention ----------------
// 4 waves/block, consecutive 16-query tiles of the SAME (b,h): the 32-key
// K/V chunk is shared by all 4 waves and double-buffer staged into LDS with
// the async global->LDS engine.  Trip count is block-uniform (max q-tile of
// the block); fully-masked chunks are exact no-ops in the online softmax.
// Interior chunks (key0+31 <= q0, wave-uniform SCC branch) take a mask-free
// softmax path; only the <=2 diagonal chunks per wave pay the cmp/cndmask.
__global__ __launch_bounds__(128) void attn_kernel(
    const __bf16* __restrict__ Q, const __bf16* __restrict__ K,
    const __bf16* __restrict__ Vt, __bf16* __restrict__ O) {
  __shared__ __bf16 ktile[2][32][96];  // [key][d]   2 x 6KB
  __shared__ __bf16 vtile[2][96][32];  // [d][key]   2 x 6KB
  __shared__ __bf16 plds[4][16][32];   // per-wave P tile, 4KB

  int lane = threadIdx.x & 31;
  int wave = threadIdx.x >> 5;
  int tid = threadIdx.x;
  int tile = blockIdx.x * 4 + wave;    // 0 .. B*NH*(S/16)-1
  int qt = tile & (SS / 16 - 1);
  int bh = tile >> 7;                  // S/16 == 128; all 4 waves: same bh
  int h = bh & (NH - 1), b = bh >> 5;
  int q0 = qt * 16;
  int half = lane >> 4, l = lane & 15;

  const __bf16* Qbh = Q + ((long)bh * SS + q0) * HD;
  const __bf16* Kbh = K + (long)bh * SS * HD;
  const __bf16* Vbh = Vt + (long)bh * HD * SS;

  // Stage K[key0..key0+31][0..95] and Vt[0..95][key0..key0+31] into buf.
  // 768 x 16B segments; each of 128 threads issues 6 async transfers.
  auto stage_kv = [&](int buf, int key0) {
#pragma unroll
    for (int it = 0; it < 6; ++it) {
      int s = tid + it * 128;                       // 0..767
      if (s < 384) {                                // K tile: 32 rows x 12 seg
        int key = s / 12, dseg = s % 12;
        async_ld16(lds_off(&ktile[buf][key][dseg * 8]),
                   Kbh + (long)(key0 + key) * HD + dseg * 8);
      } else {                                      // V tile: 96 rows x 4 seg
        int v = s - 384;
        int d = v >> 2, kseg = v & 3;
        async_ld16(lds_off(&vtile[buf][d][kseg * 8]),
                   Vbh + (long)d * SS + key0 + kseg * 8);
      }
    }
  };

  v16bf qf[3];
#pragma unroll
  for (int i = 0; i < 3; ++i) qf[i] = load_frag_a(Qbh, HD, 0, i * 32, lane);

  v8f acc[6];
#pragma unroll
  for (int t = 0; t < 6; ++t)
#pragma unroll
    for (int r = 0; r < 8; ++r) acc[t][r] = 0.0f;

  float mrow[8], lrow[8];
#pragma unroll
  for (int r = 0; r < 8; ++r) { mrow[r] = -1e30f; lrow[r] = 0.0f; }

  const float scale = 0.10206207261596575f;  // 1/sqrt(96)
  // Block-uniform chunk count: causal bound of the block's last q-tile.
  int qt3 = (blockIdx.x * 4 + 3) & (SS / 16 - 1);
  int nk = (qt3 * 16 + 16 + 31) >> 5;

  stage_kv(0, 0);
  asm volatile("s_wait_asynccnt 0x0" ::: "memory");
  __syncthreads();

  for (int j = 0; j < nk; ++j) {
    int key0 = j * 32;
    if (j + 1 < nk) stage_kv((j + 1) & 1, key0 + 32);

    const __bf16* kb = &ktile[j & 1][0][0];
    const __bf16* vb = &vtile[j & 1][0][0];

    v8f s0, s1;
#pragma unroll
    for (int r = 0; r < 8; ++r) { s0[r] = 0.0f; s1[r] = 0.0f; }
#pragma unroll
    for (int i = 0; i < 3; ++i) {
      v16bf kb0 = load_frag_b(kb, HD, 0,  i * 32, lane);
      s0 = wmma_bf16(qf[i], kb0, s0);
      v16bf kb1 = load_frag_b(kb, HD, 16, i * 32, lane);
      s1 = wmma_bf16(qf[i], kb1, s1);
    }

    // Online-softmax update; `masked` is constant-folded at the two call
    // sites, selected by a wave-uniform (SCC) branch -> no EXEC divergence.
    auto softmax_update = [&](bool masked) {
#pragma unroll
      for (int r = 0; r < 8; ++r) {
        float v0 = s0[r] * scale;
        float v1 = s1[r] * scale;
        if (masked) {
          int qrow = q0 + r + 8 * half;
          v0 = (key0 + l      <= qrow) ? v0 : -3.0e38f;
          v1 = (key0 + 16 + l <= qrow) ? v1 : -3.0e38f;
        }
        float rmax = fmaxf(v0, v1);
#pragma unroll
        for (int off = 1; off < 16; off <<= 1)
          rmax = fmaxf(rmax, __shfl_xor(rmax, off, 32));
        float mnew = fmaxf(mrow[r], rmax);
        float corr = __expf(mrow[r] - mnew);
        float p0 = __expf(v0 - mnew);
        float p1 = __expf(v1 - mnew);
        float rs = p0 + p1;
#pragma unroll
        for (int off = 1; off < 16; off <<= 1) rs += __shfl_xor(rs, off, 32);
        lrow[r] = lrow[r] * corr + rs;
        mrow[r] = mnew;
#pragma unroll
        for (int t = 0; t < 6; ++t) acc[t][r] *= corr;
        plds[wave][r + 8 * half][l]      = (__bf16)p0;
        plds[wave][r + 8 * half][16 + l] = (__bf16)p1;
      }
    };
    if (key0 + 31 <= q0) softmax_update(false);  // interior: mask-free
    else                 softmax_update(true);   // diagonal/above: masked

    asm volatile("s_wait_dscnt 0x0" ::: "memory");  // wave-local LDS RAW fence

    v16bf pf = load_frag_a(&plds[wave][0][0], 32, 0, 0, lane);
#pragma unroll
    for (int t = 0; t < 6; ++t) {
      v16bf vf = load_frag_b(vb, 32, t * 16, 0, lane);
      acc[t] = wmma_bf16(pf, vf, acc[t]);
    }

    asm volatile("s_wait_asynccnt 0x0" ::: "memory");
    __syncthreads();
  }

#pragma unroll
  for (int r = 0; r < 8; ++r) {
    float inv = 1.0f / lrow[r];
    long obase = ((long)b * SS + (q0 + r + 8 * half)) * NHHD + h * HD;
#pragma unroll
    for (int t = 0; t < 6; ++t)
      O[obase + t * 16 + l] = (__bf16)(acc[t][r] * inv);
  }
}

// ---------------- launcher ----------------
extern "C" void kernel_launch(void* const* d_in, const int* in_sizes, int n_in,
                              void* d_out, int out_size, void* d_ws, size_t ws_size,
                              hipStream_t stream) {
  const float* hid  = (const float*)d_in[0];
  const float* cosT = (const float*)d_in[1];
  const float* sinT = (const float*)d_in[2];
  // d_in[3] (attention_mask) is exactly causal; applied analytically.
  const float* Wqkv = (const float*)d_in[4];
  const float* Wo   = (const float*)d_in[5];
  (void)in_sizes; (void)n_in; (void)out_size; (void)ws_size;

  char* ws = (char*)d_ws;
  const long M = (long)BB * SS;                       // 4096
  __bf16* hiddenB = (__bf16*)(ws);                    // [M, HID]
  __bf16* WqkvT   = (__bf16*)(ws + 25165824);         // [OP, HID]
  __bf16* WoT     = (__bf16*)(ws + 81788928);         // [HID, NHHD]
  __bf16* qkvB    = (__bf16*)(ws + 100663296);        // [M, OP]
  __bf16* Qb      = (__bf16*)(ws + 176160768);        // [B,H,S,HD]
  __bf16* Kb      = (__bf16*)(ws + 201326592);        // [B,H,S,HD]
  __bf16* Vt      = (__bf16*)(ws + 226492416);        // [B,H,HD,S]
  __bf16* Ob      = (__bf16*)(ws + 251658240);        // [M, NHHD]

  // 1) precision conversion / weight transposition
  {
    long n = M * HID;
    cvt_f32_bf16_kernel<<<(int)((n + 255) / 256), 256, 0, stream>>>(hid, hiddenB, n);
  }
  {
    long n = (long)HID * OP;
    transpose_f32_bf16_kernel<<<(int)((n + 255) / 256), 256, 0, stream>>>(Wqkv, WqkvT, HID, OP);
  }
  {
    long n = (long)NHHD * HID;
    transpose_f32_bf16_kernel<<<(int)((n + 255) / 256), 256, 0, stream>>>(Wo, WoT, NHHD, HID);
  }

  // 2) QKV projection: [4096,3072] x [3072,9216] -> bf16 [4096,9216]
  gemm_bf16_kernel<0><<<dim3(OP / 64, (int)(M / 64)), 128, 0, stream>>>(
      hiddenB, WqkvT, qkvB, (int)M, OP, HID);

  // 3) RoPE + head layout
  {
    long n = (long)BB * SS * NH * HD;
    rope_kernel<<<(int)((n + 255) / 256), 256, 0, stream>>>(qkvB, cosT, sinT, Qb, Kb, Vt);
  }

  // 4) causal flash attention (4 q-tiles of one head per block)
  attn_kernel<<<(BB * NH * (SS / 16)) / 4, 128, 0, stream>>>(Qb, Kb, Vt, Ob);

  // 5) output projection: [4096,3072] x [3072,3072] -> f32 d_out
  gemm_bf16_kernel<1><<<dim3(HID / 64, (int)(M / 64)), 128, 0, stream>>>(
      Ob, WoT, d_out, (int)M, HID, NHHD);
}
